// GaussianPolicy_30743375904785
// MI455X (gfx1250) — matrix operand order repaired and checked by reading, not compiled, not run on hardware
//
#include <hip/hip_runtime.h>
#include <hip/hip_bf16.h>

typedef __attribute__((ext_vector_type(16))) __bf16 v16bf;
typedef __attribute__((ext_vector_type(8)))  float  v8f;
typedef __attribute__((ext_vector_type(4)))  float  v4f;

#define HDIM 128
#define ADIM 8
// LDS B-fragment store: 8 fragments x 32 lanes x 12 dwords (8 valid + 4 pad)
#define BSTRIDE 12
#define BFRAG_DW (8 * 32 * BSTRIDE)
#define TILES_PER_WAVE 12

__device__ __forceinline__ __bf16 tobf(float x) { return (__bf16)x; }

__device__ __forceinline__ unsigned int pack_bf16(float lo, float hi) {
    union { __bf16 b[2]; unsigned int u; } r;
    r.b[0] = (__bf16)lo; r.b[1] = (__bf16)hi;
    return r.u;
}

__device__ __forceinline__ v16bf load_bfrag(const unsigned int* __restrict__ Bsh,
                                            int nt, int lane) {
    union { uint4 q[2]; v16bf v; } u;
    const uint4* p = (const uint4*)(Bsh + nt * (32 * BSTRIDE) + lane * BSTRIDE);
    u.q[0] = p[0];
    u.q[1] = p[1];
    return u.v;
}

// ---------------------------------------------------------------------------
__global__ void zero_ws_kernel(float* __restrict__ p, int n) {
    int i = blockIdx.x * blockDim.x + threadIdx.x;
    if (i < n) p[i] = 0.0f;
}

// ---------------------------------------------------------------------------
// gn_pre[g,h] = bn1[h] + sum_k global[g,k] * Wgn1[k,h]   (tiny: 1024x128, K=16)
__global__ void gn_precompute_kernel(const float* __restrict__ gfeat,
                                     const float* __restrict__ Wgn,
                                     const float* __restrict__ bn,
                                     float* __restrict__ gn_pre, int total) {
    int i = blockIdx.x * blockDim.x + threadIdx.x;
    if (i >= total) return;
    int g = i >> 7, h = i & 127;
    float s = bn[h];
    #pragma unroll
    for (int k = 0; k < 16; ++k) s += gfeat[g * 16 + k] * Wgn[k * HDIM + h];
    gn_pre[i] = s;
}

// ---------------------------------------------------------------------------
// Edge pass: e1 = relu([ef | global[g]] @ [We;Wge] + be), segment-sum into esum.
// One wave owns a contiguous block of 16-row tiles; per-graph partials are
// register-accumulated (acc[nt]) and flushed with atomics only on graph change.
__global__ void __launch_bounds__(256)
edge_pass_kernel(const float* __restrict__ ef,
                 const float* __restrict__ gfeat,
                 const int*   __restrict__ gidx,
                 const float* __restrict__ We,
                 const float* __restrict__ Wge,
                 const float* __restrict__ be,
                 float* __restrict__ esum, int NEc, int nGraphs) {
    __shared__ __align__(16) unsigned int Bsh[BFRAG_DW];

    // Stacked B = [We (K0..15) ; Wge (K16..31)], 32x128, in WMMA lane layout:
    // lane half h: dwords 0-3 hold K=8h..8h+7, dwords 4-7 hold K=16+8h..16+8h+7 (table order).
    for (int s = threadIdx.x; s < 8 * 32 * 8; s += blockDim.x) {
        int d    = s & 7;
        int lane = (s >> 3) & 31;
        int nt   = s >> 8;
        int ncol = nt * 16 + (lane & 15);
        const float* Wsrc = (d >= 4) ? Wge : We;
        int i0 = ((lane >= 16) ? 8 : 0) + 2 * (d & 3);
        Bsh[nt * (32 * BSTRIDE) + lane * BSTRIDE + d] =
            pack_bf16(Wsrc[i0 * HDIM + ncol], Wsrc[(i0 + 1) * HDIM + ncol]);
    }
    __syncthreads();

    const int lane  = threadIdx.x & 31;
    const bool hi   = lane >= 16;
    const int  nlo  = lane & 15;
    const int  wavesPerBlock = blockDim.x >> 5;
    const int  nTiles = (NEc + 15) >> 4;
    const int  akb   = hi ? 8 : 0;   // A K-base within each 16-wide half
    const int  mbase = hi ? 8 : 0;   // C/D M-base

    float cbias[8];
    #pragma unroll
    for (int nt = 0; nt < 8; ++nt) cbias[nt] = be[nt * 16 + nlo];

    // blocked tile range for this wave
    const int waveId = blockIdx.x * wavesPerBlock + (threadIdx.x >> 5);
    const int totalWaves = gridDim.x * wavesPerBlock;
    const int tpw = (nTiles + totalWaves - 1) / totalWaves;
    int tStart = waveId * tpw;
    int tEnd   = tStart + tpw; if (tEnd > nTiles) tEnd = nTiles;

    float acc[8];
    #pragma unroll
    for (int nt = 0; nt < 8; ++nt) acc[nt] = 0.0f;
    int curG = -1;   // scalar: graph currently accumulated in acc

    for (int tile = tStart; tile < tEnd; ++tile) {
        const int rowBase = tile << 4;
        int row  = rowBase + nlo;
        int rowc = row < NEc ? row : NEc - 1;
        const int g = gidx[rowc];

        // A 16x32: K0..15 = edge feats (streamed NT), K16..31 = global feats
        const v4f* e4 = (const v4f*)(ef + (size_t)rowc * 16 + akb);
        v4f e0 = __builtin_nontemporal_load(e4);
        v4f e1 = __builtin_nontemporal_load(e4 + 1);
        const float* gl = gfeat + (size_t)g * 16;
        v16bf a;
        #pragma unroll
        for (int i = 0; i < 4; ++i) {
            a[i]      = tobf(e0[i]);
            a[4 + i]  = tobf(e1[i]);
            a[8 + i]  = tobf(gl[akb + i]);
            a[12 + i] = tobf(gl[akb + 4 + i]);
        }

        const bool full = (rowBase + 16) <= NEc;
        const int  g0   = gidx[rowBase];
        const int  g15  = gidx[(rowBase + 15) < NEc ? (rowBase + 15) : (NEc - 1)];
        // scalarize: branches below are s_cbranch, EXEC stays all-1s for WMMA
        const int uni = __builtin_amdgcn_readfirstlane((full && (g0 == g15)) ? 1 : 0);
        const int g0s = __builtin_amdgcn_readfirstlane(g0);

        if (uni) {
            if (g0s != curG) {
                if (curG >= 0) {
                    #pragma unroll
                    for (int nt = 0; nt < 8; ++nt)
                        atomicAdd(&esum[(size_t)curG * HDIM + nt * 16 + nlo], acc[nt]);
                }
                #pragma unroll
                for (int nt = 0; nt < 8; ++nt) acc[nt] = 0.0f;
                curG = g0s;
            }
            #pragma unroll
            for (int nt = 0; nt < 8; ++nt) {
                v16bf b = load_bfrag(Bsh, nt, lane);
                v8f c;
                #pragma unroll
                for (int j = 0; j < 8; ++j) c[j] = cbias[nt];
                v8f d = __builtin_amdgcn_wmma_f32_16x16x32_bf16(
                            false, a, false, b, (short)0, c, false, false);
                float s = 0.0f;
                #pragma unroll
                for (int j = 0; j < 8; ++j) s += fmaxf(d[j], 0.0f);
                acc[nt] += s;
            }
        } else {
            // rare: mixed-graph or tail tile -> direct atomics, OOB rows to dump
            int grows[8];
            #pragma unroll
            for (int j = 0; j < 8; ++j) {
                int mr  = rowBase + mbase + j;
                int mrc = mr < NEc ? mr : NEc - 1;
                int gv  = gidx[mrc];
                grows[j] = (mr < NEc) ? gv : nGraphs;
            }
            #pragma unroll
            for (int nt = 0; nt < 8; ++nt) {
                v16bf b = load_bfrag(Bsh, nt, lane);
                v8f c;
                #pragma unroll
                for (int j = 0; j < 8; ++j) c[j] = cbias[nt];
                v8f d = __builtin_amdgcn_wmma_f32_16x16x32_bf16(
                            false, a, false, b, (short)0, c, false, false);
                const int ncol = nt * 16 + nlo;
                #pragma unroll
                for (int j = 0; j < 8; ++j)
                    atomicAdd(&esum[(size_t)grows[j] * HDIM + ncol], fmaxf(d[j], 0.0f));
            }
        }
    }
    if (curG >= 0) {
        #pragma unroll
        for (int nt = 0; nt < 8; ++nt)
            atomicAdd(&esum[(size_t)curG * HDIM + nt * 16 + nlo], acc[nt]);
    }
}

// ---------------------------------------------------------------------------
// Node pass: n1 = relu(nf @ Wn1 + gn_pre[g]); same accumulation scheme.
__global__ void __launch_bounds__(256)
node_pass_kernel(const float* __restrict__ nf,
                 const int*   __restrict__ gidx,
                 const float* __restrict__ Wn,
                 const float* __restrict__ gn_pre,
                 float* __restrict__ nsum, int NNc, int nGraphs) {
    __shared__ __align__(16) unsigned int Bsh[BFRAG_DW];

    for (int s = threadIdx.x; s < 8 * 32 * 8; s += blockDim.x) {
        int d    = s & 7;
        int lane = (s >> 3) & 31;
        int nt   = s >> 8;
        int ncol = nt * 16 + (lane & 15);
        int kb   = (lane >= 16) ? 8 : 0;
        int i0   = 2 * d + (d >= 4 ? 8 : 0);
        Bsh[nt * (32 * BSTRIDE) + lane * BSTRIDE + d] =
            pack_bf16(Wn[(kb + i0) * HDIM + ncol], Wn[(kb + i0 + 1) * HDIM + ncol]);
    }
    __syncthreads();

    const int lane  = threadIdx.x & 31;
    const bool hi   = lane >= 16;
    const int  nlo  = lane & 15;
    const int  wavesPerBlock = blockDim.x >> 5;
    const int  nTiles = (NNc + 15) >> 4;
    const int  akb   = hi ? 8 : 0;
    const int  mbase = hi ? 8 : 0;

    const int waveId = blockIdx.x * wavesPerBlock + (threadIdx.x >> 5);
    const int totalWaves = gridDim.x * wavesPerBlock;
    const int tpw = (nTiles + totalWaves - 1) / totalWaves;
    int tStart = waveId * tpw;
    int tEnd   = tStart + tpw; if (tEnd > nTiles) tEnd = nTiles;

    float acc[8];
    #pragma unroll
    for (int nt = 0; nt < 8; ++nt) acc[nt] = 0.0f;
    int curG = -1;

    for (int tile = tStart; tile < tEnd; ++tile) {
        const int rowBase = tile << 4;
        int row  = rowBase + nlo;
        int rowc = row < NNc ? row : NNc - 1;

        const float* r = nf + (size_t)rowc * 32;
        v4f n0 = __builtin_nontemporal_load((const v4f*)(r + akb));
        v4f n1 = __builtin_nontemporal_load((const v4f*)(r + akb + 4));
        v4f n2 = __builtin_nontemporal_load((const v4f*)(r + 16 + akb));
        v4f n3 = __builtin_nontemporal_load((const v4f*)(r + 16 + akb + 4));
        v16bf a;
        #pragma unroll
        for (int i = 0; i < 4; ++i) {
            a[i]      = tobf(n0[i]);
            a[4 + i]  = tobf(n1[i]);
            a[8 + i]  = tobf(n2[i]);
            a[12 + i] = tobf(n3[i]);
        }

        const bool full = (rowBase + 16) <= NNc;
        const int  g0   = gidx[rowBase];
        const int  g15  = gidx[(rowBase + 15) < NNc ? (rowBase + 15) : (NNc - 1)];
        const int uni = __builtin_amdgcn_readfirstlane((full && (g0 == g15)) ? 1 : 0);
        const int g0s = __builtin_amdgcn_readfirstlane(g0);

        if (uni) {
            if (g0s != curG) {
                if (curG >= 0) {
                    #pragma unroll
                    for (int nt = 0; nt < 8; ++nt)
                        atomicAdd(&nsum[(size_t)curG * HDIM + nt * 16 + nlo], acc[nt]);
                }
                #pragma unroll
                for (int nt = 0; nt < 8; ++nt) acc[nt] = 0.0f;
                curG = g0s;
            }
            const float* gp = gn_pre + (size_t)g0s * HDIM;
            #pragma unroll
            for (int nt = 0; nt < 8; ++nt) {
                v16bf b = load_bfrag(Bsh, nt, lane);
                const float cv = gp[nt * 16 + nlo];
                v8f c;
                #pragma unroll
                for (int j = 0; j < 8; ++j) c[j] = cv;
                v8f d = __builtin_amdgcn_wmma_f32_16x16x32_bf16(
                            false, a, false, b, (short)0, c, false, false);
                float s = 0.0f;
                #pragma unroll
                for (int j = 0; j < 8; ++j) s += fmaxf(d[j], 0.0f);
                acc[nt] += s;
            }
        } else {
            int grows[8];
            #pragma unroll
            for (int j = 0; j < 8; ++j) {
                int mr  = rowBase + mbase + j;
                int mrc = mr < NNc ? mr : NNc - 1;
                int gv  = gidx[mrc];
                grows[j] = (mr < NNc) ? gv : nGraphs;
            }
            #pragma unroll
            for (int nt = 0; nt < 8; ++nt) {
                v16bf b = load_bfrag(Bsh, nt, lane);
                const int ncol = nt * 16 + nlo;
                v8f c;
                #pragma unroll
                for (int j = 0; j < 8; ++j) {
                    int gj = grows[j] < nGraphs ? grows[j] : 0;  // dump rows: C irrelevant
                    c[j] = gn_pre[(size_t)gj * HDIM + ncol];
                }
                v8f d = __builtin_amdgcn_wmma_f32_16x16x32_bf16(
                            false, a, false, b, (short)0, c, false, false);
                #pragma unroll
                for (int j = 0; j < 8; ++j)
                    atomicAdd(&nsum[(size_t)grows[j] * HDIM + ncol], fmaxf(d[j], 0.0f));
            }
        }
    }
    if (curG >= 0) {
        #pragma unroll
        for (int nt = 0; nt < 8; ++nt)
            atomicAdd(&nsum[(size_t)curG * HDIM + nt * 16 + nlo], acc[nt]);
    }
}

// ---------------------------------------------------------------------------
__device__ __forceinline__ int lower_bound_i(const int* __restrict__ a, int n, int v) {
    int lo = 0, hi = n;
    while (lo < hi) { int mid = (lo + hi) >> 1; if (a[mid] < v) lo = mid + 1; else hi = mid; }
    return lo;
}

// One block per graph: counts via binary search on sorted idx arrays, then the
// two tiny [128]->[8] heads.
__global__ void finalize_kernel(const float* __restrict__ nsum,
                                const float* __restrict__ esum,
                                const int*   __restrict__ ngidx,
                                const int*   __restrict__ egidx,
                                const float* __restrict__ Wnm, const float* __restrict__ Wem,
                                const float* __restrict__ bm,
                                const float* __restrict__ Wns, const float* __restrict__ Wes,
                                const float* __restrict__ bs,
                                float* __restrict__ out, int NNc, int NEc, int lsOff) {
    const int g = blockIdx.x;
    const int t = threadIdx.x;   // 128 threads
    __shared__ float na[HDIM], ea[HDIM];
    __shared__ int cnts[2];
    if (t == 0) cnts[0] = lower_bound_i(ngidx, NNc, g + 1) - lower_bound_i(ngidx, NNc, g);
    if (t == 1) cnts[1] = lower_bound_i(egidx, NEc, g + 1) - lower_bound_i(egidx, NEc, g);
    __syncthreads();
    const float ninv = 1.0f / fmaxf((float)cnts[0], 1.0f);
    const float einv = 1.0f / fmaxf((float)cnts[1], 1.0f);
    na[t] = nsum[(size_t)g * HDIM + t] * ninv;
    ea[t] = esum[(size_t)g * HDIM + t] * einv;
    __syncthreads();
    if (t < ADIM) {
        float s = bm[t];
        for (int h = 0; h < HDIM; ++h) s += na[h] * Wnm[h * ADIM + t] + ea[h] * Wem[h * ADIM + t];
        out[g * ADIM + t] = s;
    } else if (t < 2 * ADIM) {
        int a = t - ADIM;
        float s = bs[a];
        for (int h = 0; h < HDIM; ++h) s += na[h] * Wns[h * ADIM + a] + ea[h] * Wes[h * ADIM + a];
        out[lsOff + g * ADIM + a] = fminf(fmaxf(s, -20.0f), 2.0f);
    }
}

// ---------------------------------------------------------------------------
extern "C" void kernel_launch(void* const* d_in, const int* in_sizes, int n_in,
                              void* d_out, int out_size, void* d_ws, size_t ws_size,
                              hipStream_t stream) {
    const float* node_feats   = (const float*)d_in[0];
    const float* edge_feats   = (const float*)d_in[1];
    const float* global_feats = (const float*)d_in[2];
    const int*   ngidx        = (const int*)d_in[3];
    const int*   egidx        = (const int*)d_in[4];
    const float* We1  = (const float*)d_in[5];
    const float* Wge1 = (const float*)d_in[6];
    const float* be1  = (const float*)d_in[7];
    const float* Wn1  = (const float*)d_in[8];
    const float* Wgn1 = (const float*)d_in[9];
    const float* bn1  = (const float*)d_in[10];
    const float* Wnm  = (const float*)d_in[11];
    const float* Wem  = (const float*)d_in[12];
    const float* bm   = (const float*)d_in[13];
    const float* Wns  = (const float*)d_in[14];
    const float* Wes  = (const float*)d_in[15];
    const float* bs   = (const float*)d_in[16];

    const int NN = in_sizes[0] / 32;
    const int NE = in_sizes[1] / 16;
    const int B  = in_sizes[2] / 16;
    float* out = (float*)d_out;

    // workspace layout: [esum (B+1)*128][nsum (B+1)*128][gn_pre B*128]
    // (row B of each accumulator is the dumpster for out-of-range tile rows)
    float* esum   = (float*)d_ws;
    float* nsum   = esum + (size_t)(B + 1) * HDIM;
    float* gn_pre = nsum + (size_t)(B + 1) * HDIM;

    const int accTotal = 2 * (B + 1) * HDIM;
    zero_ws_kernel<<<(accTotal + 255) / 256, 256, 0, stream>>>(esum, accTotal);

    const int gnTotal = B * HDIM;
    gn_precompute_kernel<<<(gnTotal + 255) / 256, 256, 0, stream>>>(
        global_feats, Wgn1, bn1, gn_pre, gnTotal);

    const int wavesPerBlock = 8;                 // 256 threads = 8 wave32
    const int eTiles = (NE + 15) / 16;
    const int nTiles = (NN + 15) / 16;
    const int tilesPerBlock = wavesPerBlock * TILES_PER_WAVE;
    int eBlocks = (eTiles + tilesPerBlock - 1) / tilesPerBlock;
    int nBlocks = (nTiles + tilesPerBlock - 1) / tilesPerBlock;
    if (eBlocks < 1) eBlocks = 1;
    if (nBlocks < 1) nBlocks = 1;

    edge_pass_kernel<<<eBlocks, 256, 0, stream>>>(
        edge_feats, global_feats, egidx, We1, Wge1, be1, esum, NE, B);
    node_pass_kernel<<<nBlocks, 256, 0, stream>>>(
        node_feats, ngidx, Wn1, gn_pre, nsum, NN, B);

    finalize_kernel<<<B, HDIM, 0, stream>>>(
        nsum, esum, ngidx, egidx, Wnm, Wem, bm, Wns, Wes, bs,
        out, NN, NE, out_size / 2);
}
